// PrunableLeViTAttentionSubsample_33947421507745
// MI455X (gfx1250) — matrix operand order repaired and checked by reading, not compiled, not run on hardware
//
#include <hip/hip_runtime.h>
#include <hip/hip_fp16.h>

typedef _Float16 h16 __attribute__((ext_vector_type(16)));
typedef _Float16 h8v __attribute__((ext_vector_type(8)));
typedef float    f8v __attribute__((ext_vector_type(8)));
typedef uint32_t u32x4 __attribute__((ext_vector_type(4)));
typedef uint32_t u32x8 __attribute__((ext_vector_type(8)));

#define LN_EPS 1e-5f
#define QK_SCALE 0.17677669529663687f   // 32^-0.5

// ---------------------------------------------------------------------------
// WMMA fragment loaders from LDS (strides in halves).
// A operand (16xK=32 f16): lane<16 -> M=lane, K {k0..k0+7, k0+16..k0+23};
//                          lane>=16 -> M=lane-16, K {k0+8..15, k0+24..31}.
// B operand (K=32 x 16 f16): lane<16 -> N=lane, K {k0..k0+7, k0+16..k0+23};
//                            lane>=16 -> N=lane-16, K {k0+8..15, k0+24..31}.
// ---------------------------------------------------------------------------
__device__ __forceinline__ h16 frag_a(const _Float16* base, int ld) {
  int lane = threadIdx.x & 31;
  int m = lane & 15, hi = lane >> 4;
  const _Float16* p = base + m * ld + hi * 8;
  h8v lo = *(const h8v*)p;
  h8v hiP = *(const h8v*)(p + 16);
  h16 r;
#pragma unroll
  for (int i = 0; i < 8; ++i) { r[i] = lo[i]; r[i + 8] = hiP[i]; }
  return r;
}

__device__ __forceinline__ h16 frag_b(const _Float16* base, int ld) {
  int lane = threadIdx.x & 31;
  int n = lane & 15, kh = lane >> 4;
  const _Float16* p = base + n * ld + kh * 8;
  h8v x0 = *(const h8v*)p;
  h8v x1 = *(const h8v*)(p + 16);
  h16 r;
#pragma unroll
  for (int i = 0; i < 8; ++i) { r[i] = x0[i]; r[i + 8] = x1[i]; }
  return r;
}

__device__ __forceinline__ f8v wmma_f16(h16 a, h16 b, f8v c) {
  return __builtin_amdgcn_wmma_f32_16x16x32_f16(false, a, false, b, (short)0, c,
                                                false, false);
}

__device__ __forceinline__ void cvt_store8(_Float16* dst, float4 a, float4 b) {
  h8v h;
  h[0] = (_Float16)a.x; h[1] = (_Float16)a.y; h[2] = (_Float16)a.z; h[3] = (_Float16)a.w;
  h[4] = (_Float16)b.x; h[5] = (_Float16)b.y; h[6] = (_Float16)b.z; h[7] = (_Float16)b.w;
  *(h8v*)dst = h;
}

// ---------------------------------------------------------------------------
// TDM issue: 2-D f16 tile (32 halves x tile_d1 rows, row stride srow halves)
// global -> LDS, hardware-padded +4 DWORDs after every 16 DWORDs so LDS rows
// land at stride 40 halves (matches the frag loaders / bank padding).
// Descriptor per CDNA5 ISA.
// ---------------------------------------------------------------------------
__device__ __forceinline__ void tdm_load_2d_f16(uint32_t lds_off,
                                                const _Float16* gptr,
                                                uint32_t tile_d1,
                                                uint32_t srow) {
  uint64_t ga = (uint64_t)(uintptr_t)gptr;
  u32x4 g0;
  g0[0] = 1u;                              // count=1, user descriptor
  g0[1] = lds_off;                         // lds_addr (bytes)
  g0[2] = (uint32_t)ga;                    // global_addr[31:0]
  g0[3] = (uint32_t)((ga >> 32) & 0x01FFFFFFu) | (2u << 30);  // [56:32], type=2
  u32x8 g1;
  g1[0] = (1u << 16)      // data_size = 2 bytes
        | (1u << 20)      // pad_enable
        | (3u << 22)      // pad_interval: 16 DWORDs (= 64B = one 32-half row)
        | (3u << 25);     // pad_amount: 4 DWORDs (= 8 halves -> stride 40)
  g1[1] = (32u & 0xFFFFu) << 16;           // tensor_dim0 = 32
  g1[2] = (tile_d1 & 0xFFFFu) << 16;       // tensor_dim1 (dim0 hi16 = 0)
  g1[3] = 32u << 16;                       // tile_dim0 = 32 (dim1 hi16 = 0)
  g1[4] = tile_d1;                         // tile_dim1 (tile_dim2 = 0)
  g1[5] = srow;                            // tensor_dim0_stride lo32
  g1[6] = 0u;                              // stride0 hi16, stride1 lo16
  g1[7] = 0u;                              // stride1 hi32
  asm volatile("tensor_load_to_lds %0, %1" :: "s"(g0), "s"(g1) : "memory");
}

// ---------------------------------------------------------------------------
// Generic fused GEMM + layer-scale epilogue:  Y = (A @ W^T) * inv + beta
// Tile: 128(M) x 64(N) x 32(K), 8 waves, 32x32 WMMA quad per wave.
// ---------------------------------------------------------------------------
enum { A_F32 = 0, A_F32_SUB = 1, A_F16 = 2 };
enum { O_F16 = 0, O_F32 = 1 };

template <int AMODE, int OMODE>
__global__ __launch_bounds__(256) void gemm_ln(
    const float* __restrict__ A32, const _Float16* __restrict__ A16, int lda,
    const float* __restrict__ W, int K,
    const float* __restrict__ gg, const float* __restrict__ bb,
    const float* __restrict__ mm, const float* __restrict__ vv,
    float* __restrict__ O32, _Float16* __restrict__ O16, int ldo, int nblocks) {
  __shared__ _Float16 aT[128 * 40];
  __shared__ _Float16 bT[64 * 40];

  const int bm = blockIdx.x / nblocks;
  const int bn = blockIdx.x % nblocks;
  const int t = threadIdx.x;
  const int w = t >> 5;
  const int wm = w & 3, wn = w >> 2;

  const int arow = t >> 1, ahalf = t & 1;
  long asrc;
  {
    long gm = (long)bm * 128 + arow;
    if (AMODE == A_F32_SUB) {  // gather the 2-strided 7x7 query grid of x
      long bidx = gm / 49;
      int r = (int)(gm % 49);
      asrc = bidx * 196 + (r / 7) * 28 + (r % 7) * 2;
    } else {
      asrc = gm;
    }
  }
  const int brow = (t & 127) >> 1, bhalf = t & 1;

  f8v acc00 = (f8v){}, acc01 = (f8v){}, acc10 = (f8v){}, acc11 = (f8v){};

  for (int k0 = 0; k0 < K; k0 += 32) {
    {  // A tile -> LDS
      _Float16* dst = &aT[arow * 40 + ahalf * 16];
      if (AMODE == A_F16) {
        const _Float16* s = A16 + asrc * lda + k0 + ahalf * 16;
        *(h8v*)dst = *(const h8v*)s;
        *(h8v*)(dst + 8) = *(const h8v*)(s + 8);
        if (k0 + 32 < K) __builtin_prefetch((const void*)(s + 32), 0, 0);
      } else {
        const float* sp = A32 + asrc * (long)lda + k0 + ahalf * 16;
        const float4* s = (const float4*)sp;
        float4 f0 = s[0], f1 = s[1], f2 = s[2], f3 = s[3];
        cvt_store8(dst, f0, f1);
        cvt_store8(dst + 8, f2, f3);
        if (k0 + 32 < K) __builtin_prefetch((const void*)(sp + 32), 0, 0);
      }
    }
    if (t < 128) {  // W tile (N x K row-major) -> LDS
      const float* sp = W + (long)(bn * 64 + brow) * K + k0 + bhalf * 16;
      const float4* s = (const float4*)sp;
      float4 f0 = s[0], f1 = s[1], f2 = s[2], f3 = s[3];
      _Float16* dst = &bT[brow * 40 + bhalf * 16];
      cvt_store8(dst, f0, f1);
      cvt_store8(dst + 8, f2, f3);
      if (k0 + 32 < K) __builtin_prefetch((const void*)(sp + 32), 0, 0);
    }
    __syncthreads();

    h16 af0 = frag_a(&aT[(wm * 32) * 40], 40);
    h16 af1 = frag_a(&aT[(wm * 32 + 16) * 40], 40);
    h16 bf0 = frag_b(&bT[(wn * 32) * 40], 40);
    h16 bf1 = frag_b(&bT[(wn * 32 + 16) * 40], 40);
    acc00 = wmma_f16(af0, bf0, acc00);
    acc01 = wmma_f16(af0, bf1, acc01);
    acc10 = wmma_f16(af1, bf0, acc10);
    acc11 = wmma_f16(af1, bf1, acc11);
    __syncthreads();
  }

  const int lane = t & 31;
  const int cn = lane & 15, hi = lane >> 4;
#pragma unroll
  for (int sn = 0; sn < 2; ++sn) {
    int n = bn * 64 + wn * 32 + sn * 16 + cn;
    float inv = gg[n] * rsqrtf(vv[n] + LN_EPS);
    float beta = bb[n] - mm[n] * inv;
#pragma unroll
    for (int sm = 0; sm < 2; ++sm) {
      const f8v& c = (sm == 0) ? (sn == 0 ? acc00 : acc01)
                               : (sn == 0 ? acc10 : acc11);
      long m0 = (long)bm * 128 + wm * 32 + sm * 16 + hi * 8;
#pragma unroll
      for (int j = 0; j < 8; ++j) {
        float val = c[j] * inv + beta;
        long row = m0 + j;
        if (OMODE == O_F16) O16[row * ldo + n] = (_Float16)val;
        else                O32[row * ldo + n] = val;
      }
    }
  }
}

// ---------------------------------------------------------------------------
// Fused attention per (batch, head). q/k head tiles arrive via TDM with
// hardware pad to stride 40; v is transposed manually; softmax is one wave
// per row using wave32 shfl_xor reductions; bias is staged in LDS and the
// score epilogue is fully branchless (bias value is arithmetically live on
// both select arms so the LDS load cannot be sunk into a guarded block).
// ---------------------------------------------------------------------------
#define AT_N 196
#define AT_NP 224

__global__ __launch_bounds__(256) void attn_kernel(
    const _Float16* __restrict__ kv,  // [B][196][768] f16
    const _Float16* __restrict__ q,   // [B][49][256] f16
    const float* __restrict__ bias,   // [8][196]
    _Float16* __restrict__ hs) {      // [B][49][512] f16 (hardswish'd)
  extern __shared__ char smem[];
  _Float16* q_l = (_Float16*)smem;               // 64 x 32 @ stride 40 : 5120 B
  _Float16* k_l = (_Float16*)(smem + 5120);      // 208 x 32 @ stride 40: 16640 B
  _Float16* vt  = (_Float16*)(smem + 21760);     // 64 x 224 (v^T)      : 28672 B
  float*    s_l = (float*)(smem + 50432);        // 64 x 208 f32        : 53248 B
  _Float16* a_l = (_Float16*)(smem + 103680);    // 64 x 224 f16        : 28672 B
  float*    bias_l = (float*)(smem + 132352);    // 196 f32             : 784 B

  const int b = blockIdx.x >> 3;
  const int h = blockIdx.x & 7;
  const int t = threadIdx.x;
  const int w = t >> 5, lane = t & 31;

  // wave 0: kick off TDM DMAs for q (49x32) and k (196x32) head slices
  if (w == 0) {
    tdm_load_2d_f16((uint32_t)(uintptr_t)q_l,
                    q + (long)b * 49 * 256 + h * 32, 49u, 256u);
    tdm_load_2d_f16((uint32_t)(uintptr_t)k_l,
                    kv + (long)b * 196 * 768 + h * 96, 196u, 768u);
  }

  // all threads: zero v^T and attn padding regions
  {
    uint32_t* z0 = (uint32_t*)vt;
    uint32_t* z1 = (uint32_t*)a_l;
    for (int i = t; i < 64 * AT_NP / 2; i += 256) { z0[i] = 0u; z1[i] = 0u; }
  }
  // v transposed: vt[vd][n]
  for (int idx = t; idx < AT_N * 64; idx += 256) {
    int n = idx >> 6, vd = idx & 63;
    vt[vd * AT_NP + n] = kv[(long)(b * 196 + n) * 768 + h * 96 + 32 + vd];
  }
  // stage this head's bias row in LDS
  if (t < AT_N) bias_l[t] = bias[h * 196 + t];

  if (w == 0) __builtin_amdgcn_s_wait_tensorcnt(0);
  __syncthreads();

  // ---- scores: wave w owns m-tile (w&3); n-tiles stride by 2 ----
  for (int tile = w; tile < 52; tile += 8) {
    int mt = tile & 3, nt = tile >> 2;
    h16 af = frag_a(&q_l[(mt * 16) * 40], 40);
    h16 bf = frag_b(&k_l[(nt * 16) * 40], 40);
    f8v acc = (f8v){};
    acc = wmma_f16(af, bf, acc);
    int n = nt * 16 + (lane & 15);
    int hi = lane >> 4;
    int ky = n / 14, kx = n - ky * 14;
#pragma unroll
    for (int j = 0; j < 8; ++j) {
      int m = mt * 16 + hi * 8 + j;
      int qy = (m / 7) * 2, qx = (m % 7) * 2;
      int dy = qy - ky; dy = (dy < 0) ? -dy : dy;
      int dx = qx - kx; dx = (dx < 0) ? -dx : dx;
      int bidx = dy * 14 + dx;
      bidx = (bidx < AT_N) ? bidx : 0;   // clamp inside allocation
      float lb = bias_l[bidx];
      bool valid = (n < AT_N) && (m < 49);
      // lb is live on BOTH select arms -> load stays unconditional, and the
      // invalid arm still evaluates to exactly -1e30 (lb << ulp(1e30)).
      float base = valid ? (acc[j] * QK_SCALE) : (-1e30f - lb);
      s_l[m * 208 + n] = base + lb;
    }
  }
  __syncthreads();

  // ---- softmax: one wave per query row, shfl_xor reductions (wave32) ----
  for (int r = w; r < 49; r += 8) {
    float* row = &s_l[r * 208];
    float mx = -1e30f;
    for (int n = lane; n < 208; n += 32) mx = fmaxf(mx, row[n]);
#pragma unroll
    for (int off = 16; off > 0; off >>= 1) mx = fmaxf(mx, __shfl_xor(mx, off, 32));
    float sum = 0.f;
    for (int n = lane; n < 208; n += 32) {
      float e = __expf(row[n] - mx);   // padded cols are -1e30 -> e == 0
      row[n] = e;
      sum += e;
    }
#pragma unroll
    for (int off = 16; off > 0; off >>= 1) sum += __shfl_xor(sum, off, 32);
    float rs = 1.f / sum;
    _Float16* arow = &a_l[r * AT_NP];
    for (int n = lane; n < 208; n += 32) arow[n] = (_Float16)(row[n] * rs);
  }
  __syncthreads();

  // ---- out = attn @ v : 4 m-tiles x 4 vd-tiles, K = 224 (7 WMMA steps) ----
#pragma unroll
  for (int i = 0; i < 2; ++i) {
    int tile = w * 2 + i;
    int mt = tile & 3, vtile = tile >> 2;
    f8v acc = (f8v){};
#pragma unroll
    for (int kk = 0; kk < 7; ++kk) {
      h16 af = frag_a(&a_l[(mt * 16) * AT_NP + kk * 32], AT_NP);
      h16 bf = frag_b(&vt[(vtile * 16) * AT_NP + kk * 32], AT_NP);
      acc = wmma_f16(af, bf, acc);
    }
    int vd = vtile * 16 + (lane & 15);
    int hi = lane >> 4;
#pragma unroll
    for (int j = 0; j < 8; ++j) {
      int m = mt * 16 + hi * 8 + j;
      if (m < 49) {
        float x = acc[j];
        float hsv = x * fminf(fmaxf(x + 3.f, 0.f), 6.f) * (1.f / 6.f);
        hs[(long)(b * 49 + m) * 512 + h * 64 + vd] = (_Float16)hsv;
      }
    }
  }
}

// ---------------------------------------------------------------------------
// Launch: kv GEMM -> q GEMM -> attention -> proj GEMM (serialized on stream).
// Workspace (f16): kv 50176x768 | q 12544x256 | hs 12544x512  => ~92 MB.
// ---------------------------------------------------------------------------
extern "C" void kernel_launch(void* const* d_in, const int* in_sizes, int n_in,
                              void* d_out, int out_size, void* d_ws,
                              size_t ws_size, hipStream_t stream) {
  const float* x    = (const float*)d_in[0];
  const float* kv_w = (const float*)d_in[1];
  const float* kv_g = (const float*)d_in[2];
  const float* kv_b = (const float*)d_in[3];
  const float* kv_m = (const float*)d_in[4];
  const float* kv_v = (const float*)d_in[5];
  const float* q_w  = (const float*)d_in[6];
  const float* q_g  = (const float*)d_in[7];
  const float* q_b  = (const float*)d_in[8];
  const float* q_m  = (const float*)d_in[9];
  const float* q_v  = (const float*)d_in[10];
  const float* p_w  = (const float*)d_in[11];
  const float* p_g  = (const float*)d_in[12];
  const float* p_b  = (const float*)d_in[13];
  const float* p_m  = (const float*)d_in[14];
  const float* p_v  = (const float*)d_in[15];
  const float* abia = (const float*)d_in[16];
  float* out = (float*)d_out;

  _Float16* kv_ws = (_Float16*)d_ws;
  _Float16* q_ws  = kv_ws + (size_t)256 * 196 * 768;
  _Float16* hs_ws = q_ws + (size_t)256 * 49 * 256;

  // kv = LN(x @ kv_w^T): M=50176, N=768, K=256 -> f16 ws
  gemm_ln<A_F32, O_F16><<<392 * 12, 256, 0, stream>>>(
      x, nullptr, 256, kv_w, 256, kv_g, kv_b, kv_m, kv_v, nullptr, kv_ws, 768,
      12);
  // q = LN(xq @ q_w^T): M=12544 (subsampled rows), N=256, K=256 -> f16 ws
  gemm_ln<A_F32_SUB, O_F16><<<98 * 4, 256, 0, stream>>>(
      x, nullptr, 256, q_w, 256, q_g, q_b, q_m, q_v, nullptr, q_ws, 256, 4);
  // fused attention + hardswish, one workgroup per (b,h)
  attn_kernel<<<2048, 256, 133136, stream>>>(kv_ws, q_ws, abia, hs_ws);
  // final projection: M=12544, N=384, K=512 -> f32 output
  gemm_ln<A_F16, O_F32><<<98 * 6, 256, 0, stream>>>(
      nullptr, hs_ws, 512, p_w, 512, p_g, p_b, p_m, p_v, out, nullptr, 384, 6);
}
